// SEDDAttention_7576322310222
// MI455X (gfx1250) — matrix operand order repaired and checked by reading, not compiled, not run on hardware
//
#include <hip/hip_runtime.h>
#include <hip/hip_bf16.h>
#include <cstdint>

// ---------- CDNA5 WMMA types ----------
typedef __attribute__((ext_vector_type(16))) __bf16 v16bf;
typedef __attribute__((ext_vector_type(8)))  __bf16 v8bf;
typedef __attribute__((ext_vector_type(8)))  float  v8f;

union BF16x16 { v16bf v; v8bf h[2]; __bf16 e[16]; };
union F32x8   { v8f v; float e[8]; };

#define WMMA_BF16(a, b, c) \
  __builtin_amdgcn_wmma_f32_16x16x32_bf16(false, (a), false, (b), (short)0, (c), false, false)

// Async global->LDS copy of 16 bytes per lane (CDNA5 data mover path, ASYNCcnt).
// Flat LDS aperture: low 32 bits of a generic pointer to __shared__ are the
// wave-relative LDS byte address (ISA 10.2).
__device__ __forceinline__ void async_copy_b128(const void* gptr, void* lptr) {
  unsigned lds = (unsigned)(uintptr_t)lptr;
  unsigned long long ga = (unsigned long long)(uintptr_t)gptr;
  asm volatile("global_load_async_to_lds_b128 %0, %1, off"
               :: "v"(lds), "v"(ga) : "memory");
}
__device__ __forceinline__ void wait_async0() {
  asm volatile("s_wait_asynccnt 0x0" ::: "memory");
}

// Problem constants (from reference setup_inputs)
constexpr int Bb = 4, Tt = 2048, Cc = 2048, Hh = 16, HD = 128;

// ---------------------------------------------------------------------------
// fp32 -> bf16 conversion (vectorized float4 loads)
// ---------------------------------------------------------------------------
__global__ __launch_bounds__(256) void cvt_f32_to_bf16(
    const float* __restrict__ in, __bf16* __restrict__ out, int n4)
{
  int i = blockIdx.x * 256 + threadIdx.x;
  if (i < n4) {
    float4 v = ((const float4*)in)[i];
    out[i * 4 + 0] = (__bf16)v.x;
    out[i * 4 + 1] = (__bf16)v.y;
    out[i * 4 + 2] = (__bf16)v.z;
    out[i * 4 + 3] = (__bf16)v.w;
  }
}

// ---------------------------------------------------------------------------
// C = A (MxK, row-major) * B^T (B is NxK, row-major), bf16 in, f32 accumulate.
// Block tile 128x128, 8 waves in a 2x4 grid, each wave 64x32 (4x2 WMMA tiles).
// Double-buffered LDS fed by async global->LDS loads; one barrier per K-step.
// ---------------------------------------------------------------------------
__global__ __launch_bounds__(256) void gemm_bf16_nt(
    const __bf16* __restrict__ A, const __bf16* __restrict__ Bm,
    __bf16* __restrict__ Cbf, float* __restrict__ Cf,
    int M, int N, int K)
{
  __shared__ alignas(32) __bf16 ldsA[2][128][32];  // 16 KB
  __shared__ alignas(32) __bf16 ldsB[2][128][32];  // 16 KB

  const int tid  = threadIdx.x;
  const int lane = tid & 31, wave = tid >> 5;
  const int half = lane >> 4, l16 = lane & 15;
  const int tM0 = blockIdx.y * 128, tN0 = blockIdx.x * 128;
  const int wM0 = (wave >> 2) * 64;   // 0 or 64
  const int wN0 = (wave & 3) * 32;    // 0,32,64,96

  F32x8 acc[4][2];
  for (int i = 0; i < 4; ++i)
    for (int j = 0; j < 2; ++j)
      for (int e = 0; e < 8; ++e) acc[i][j].e[e] = 0.f;

  const int ldRow = tid >> 1;          // 128 rows, 2 threads per row
  const int ldCol = (tid & 1) * 16;    // 16 bf16 (32B) per thread
  const __bf16* aPtr = &A[(size_t)(tM0 + ldRow) * K + ldCol];
  const __bf16* bPtr = &Bm[(size_t)(tN0 + ldRow) * K + ldCol];

  auto issueTile = [&](int buf, int k0) {
    // 32B per thread per matrix = two b128 async ops each
    async_copy_b128(aPtr + k0,     &ldsA[buf][ldRow][ldCol]);
    async_copy_b128(aPtr + k0 + 8, &ldsA[buf][ldRow][ldCol + 8]);
    async_copy_b128(bPtr + k0,     &ldsB[buf][ldRow][ldCol]);
    async_copy_b128(bPtr + k0 + 8, &ldsB[buf][ldRow][ldCol + 8]);
  };

  issueTile(0, 0);

  for (int k0 = 0; k0 < K; k0 += 32) {
    const int buf = (k0 >> 5) & 1;
    wait_async0();        // tile k resident in ldsA/ldsB[buf] (this wave)
    __syncthreads();      // ... and from every wave
    if (k0 + 32 < K) issueTile(buf ^ 1, k0 + 32);
    if (k0 + 64 < K) {
      __builtin_prefetch(aPtr + k0 + 64, 0, 3);
      __builtin_prefetch(bPtr + k0 + 64, 0, 3);
    }

    // A fragment (16x32 bf16): lanes 0-15 hold K 0..15; lanes 16-31 K 16..31
    v16bf af[4];
    #pragma unroll
    for (int i = 0; i < 4; ++i) {
      BF16x16 u;
      const __bf16* src = &ldsA[buf][wM0 + i * 16 + l16][0];
      u.h[0] = *(const v8bf*)(src + half * 16);
      u.h[1] = *(const v8bf*)(src + half * 16 + 8);
      af[i] = u.v;
    }
    // B fragment (32x16): lane's column = l16, K range = half*16 .. half*16+15
    v16bf bfg[2];
    #pragma unroll
    for (int j = 0; j < 2; ++j)
      bfg[j] = *(const v16bf*)&ldsB[buf][wN0 + j * 16 + l16][half * 16];

    #pragma unroll
    for (int i = 0; i < 4; ++i)
      #pragma unroll
      for (int j = 0; j < 2; ++j)
        acc[i][j].v = WMMA_BF16(af[i], bfg[j], acc[i][j].v);
  }

  // C/D layout: element v of VGPR -> row = v + 8*half, col = l16
  #pragma unroll
  for (int i = 0; i < 4; ++i) {
    const int mBase = tM0 + wM0 + i * 16 + 8 * half;
    #pragma unroll
    for (int j = 0; j < 2; ++j) {
      const int n = tN0 + wN0 + j * 16 + l16;
      #pragma unroll
      for (int vv = 0; vv < 8; ++vv) {
        const size_t off = (size_t)(mBase + vv) * N + n;
        if (Cf) Cf[off] = acc[i][j].e[vv];
        else    Cbf[off] = (__bf16)acc[i][j].e[vv];
      }
    }
  }
}

// ---------------------------------------------------------------------------
// RoPE on q, k AND v (reference rotates v too); split qkv [B,T,3,H,hd] bf16
// into Q/K/V [B,H,T,hd] bf16.
// ---------------------------------------------------------------------------
__global__ __launch_bounds__(256) void rope_split(
    const __bf16* __restrict__ qkv, const float* __restrict__ cosT,
    const float* __restrict__ sinT, __bf16* __restrict__ Q,
    __bf16* __restrict__ K, __bf16* __restrict__ V)
{
  int idx = blockIdx.x * 256 + threadIdx.x;   // one thread per rotary pair
  const int d = idx & 63;        idx >>= 6;
  const int h = idx & 15;        idx >>= 4;
  const int s = idx % 3;         idx /= 3;
  const int t = idx & (Tt - 1);  idx >>= 11;
  const int b = idx;

  const size_t base = (((size_t)(b * Tt + t) * 3 + s) * Hh + h) * HD;
  const float x1 = (float)qkv[base + d];
  const float x2 = (float)qkv[base + 64 + d];
  const float c  = cosT[t * 64 + d];
  const float sn = sinT[t * 64 + d];
  const float o1 = x1 * c + x2 * sn;
  const float o2 = -x1 * sn + x2 * c;

  __bf16* dst = (s == 0) ? Q : ((s == 1) ? K : V);
  const size_t ob = (((size_t)(b * Hh + h)) * Tt + t) * HD;
  dst[ob + d]      = (__bf16)o1;
  dst[ob + 64 + d] = (__bf16)o2;
}

// ---------------------------------------------------------------------------
// Flash attention: block = 128 q-rows (8 waves x 16), KV streamed in tiles of
// 64 rows through LDS (K via async global->LDS; V transposed through VGPRs).
// Online softmax; P re-swizzled via per-wave LDS tile.
// ---------------------------------------------------------------------------
__global__ __launch_bounds__(256) void flash_attn(
    const __bf16* __restrict__ Qm, const __bf16* __restrict__ Km,
    const __bf16* __restrict__ Vm, __bf16* __restrict__ Y)
{
  __shared__ alignas(32) __bf16 ldsK[64][HD];      // 16 KB (row-major K tile)
  __shared__ alignas(32) __bf16 ldsVt[HD][64];     // 16 KB (V transposed)
  __shared__ alignas(32) __bf16 ldsP[8][16][64];   // 16 KB (per-wave P tile)

  const int tid  = threadIdx.x;
  const int lane = tid & 31, wave = tid >> 5;
  const int half = lane >> 4, l16 = lane & 15;
  const int qTiles = Tt / 128;
  const int bh = blockIdx.x / qTiles;
  const int qt = blockIdx.x % qTiles;
  const __bf16* Qbh = Qm + (size_t)bh * Tt * HD;
  const __bf16* Kbh = Km + (size_t)bh * Tt * HD;
  const __bf16* Vbh = Vm + (size_t)bh * Tt * HD;
  const int qRow0 = qt * 128 + wave * 16;
  const float scale = 0.08838834764831845f;  // 1/sqrt(128)

  // Preload Q fragments (16 rows x 128 K) as four 16x32 A-fragments
  v16bf qf[4];
  #pragma unroll
  for (int f = 0; f < 4; ++f) {
    BF16x16 u;
    const __bf16* src = Qbh + (size_t)(qRow0 + l16) * HD + f * 32;
    u.h[0] = *(const v8bf*)(src + half * 16);
    u.h[1] = *(const v8bf*)(src + half * 16 + 8);
    qf[f] = u.v;
  }

  F32x8 o[8];
  float mrun[8], lrun[8];
  for (int dn = 0; dn < 8; ++dn)
    for (int e = 0; e < 8; ++e) o[dn].e[e] = 0.f;
  for (int vi = 0; vi < 8; ++vi) { mrun[vi] = -1e30f; lrun[vi] = 0.f; }

  for (int kv0 = 0; kv0 < Tt; kv0 += 64) {
    // K tile via async global->LDS (16B/lane per op; 32B per thread per pass)
    #pragma unroll
    for (int p = 0; p < 2; ++p) {
      const int e = tid + p * 256;
      const int r = e >> 3, c = (e & 7) * 16;
      const __bf16* g = &Kbh[(size_t)(kv0 + r) * HD + c];
      async_copy_b128(g,     &ldsK[r][c]);
      async_copy_b128(g + 8, &ldsK[r][c + 8]);
    }
    // V tile transposed through VGPRs
    #pragma unroll
    for (int p = 0; p < 2; ++p) {
      const int e = tid + p * 256;
      const int r = e >> 3, c = (e & 7) * 16;
      BF16x16 u;
      u.v = *(const v16bf*)&Vbh[(size_t)(kv0 + r) * HD + c];
      #pragma unroll
      for (int i = 0; i < 16; ++i) ldsVt[c + i][r] = u.e[i];
    }
    wait_async0();
    __syncthreads();

    // S = Q * K^T  (16 x 64): 4 N-tiles, K-loop of 4
    F32x8 s[4];
    #pragma unroll
    for (int nt = 0; nt < 4; ++nt) {
      F32x8 a;
      for (int e = 0; e < 8; ++e) a.e[e] = 0.f;
      #pragma unroll
      for (int f = 0; f < 4; ++f) {
        v16bf kb = *(const v16bf*)&ldsK[nt * 16 + l16][f * 32 + half * 16];
        a.v = WMMA_BF16(qf[f], kb, a.v);
      }
      s[nt] = a;
    }

    // Online softmax: rows live as (vgpr=vi + 8*half), cols across 16 lanes.
    #pragma unroll
    for (int vi = 0; vi < 8; ++vi) {
      float mx = -1e30f;
      #pragma unroll
      for (int nt = 0; nt < 4; ++nt) mx = fmaxf(mx, s[nt].e[vi]);
      mx *= scale;
      #pragma unroll
      for (int off = 1; off < 16; off <<= 1)
        mx = fmaxf(mx, __shfl_xor(mx, off, 16));
      const float mnew  = fmaxf(mrun[vi], mx);
      const float alpha = __expf(mrun[vi] - mnew);
      mrun[vi] = mnew;
      #pragma unroll
      for (int dn = 0; dn < 8; ++dn) o[dn].e[vi] *= alpha;
      float rs = 0.f;
      #pragma unroll
      for (int nt = 0; nt < 4; ++nt) {
        const float pv = __expf(s[nt].e[vi] * scale - mnew);
        rs += pv;
        ldsP[wave][8 * half + vi][nt * 16 + l16] = (__bf16)pv;
      }
      #pragma unroll
      for (int off = 1; off < 16; off <<= 1)
        rs += __shfl_xor(rs, off, 16);
      lrun[vi] = lrun[vi] * alpha + rs;
    }

    // Per-wave LDS RAW: DS ops stay in order within a wave; read P back as
    // A-fragments (16x64 -> two 16x32 fragments).
    v16bf pf[2];
    #pragma unroll
    for (int kk = 0; kk < 2; ++kk) {
      BF16x16 u;
      const __bf16* src = &ldsP[wave][l16][kk * 32];
      u.h[0] = *(const v8bf*)(src + half * 16);
      u.h[1] = *(const v8bf*)(src + half * 16 + 8);
      pf[kk] = u.v;
    }
    // O += P * V   (B = V^T stored [d][kv])
    #pragma unroll
    for (int dn = 0; dn < 8; ++dn)
      #pragma unroll
      for (int kk = 0; kk < 2; ++kk) {
        v16bf vf = *(const v16bf*)&ldsVt[dn * 16 + l16][kk * 32 + half * 16];
        o[dn].v = WMMA_BF16(pf[kk], vf, o[dn].v);
      }
    __syncthreads();
  }

  // Normalize and write Y in [B,T,H,hd] (== [B,T,C]) bf16
  const int b = bh >> 4, h = bh & 15;
  #pragma unroll
  for (int vi = 0; vi < 8; ++vi) {
    const float inv = 1.f / lrun[vi];
    const int m = qRow0 + 8 * half + vi;
    #pragma unroll
    for (int dn = 0; dn < 8; ++dn) {
      const int d = dn * 16 + l16;
      Y[(((size_t)(b * Tt + m)) * Hh + h) * HD + d] = (__bf16)(o[dn].e[vi] * inv);
    }
  }
}

// ---------------------------------------------------------------------------
extern "C" void kernel_launch(void* const* d_in, const int* in_sizes, int n_in,
                              void* d_out, int out_size, void* d_ws, size_t ws_size,
                              hipStream_t stream) {
  (void)in_sizes; (void)n_in; (void)out_size; (void)ws_size;
  const float* x    = (const float*)d_in[0];
  const float* cosT = (const float*)d_in[1];
  const float* sinT = (const float*)d_in[2];
  const float* Wqkv = (const float*)d_in[3];
  const float* Wo   = (const float*)d_in[4];
  float* out = (float*)d_out;

  auto al = [](size_t v) { return (v + 255) & ~(size_t)255; };
  char* p = (char*)d_ws;
  __bf16* xb   = (__bf16*)p; p += al((size_t)Bb * Tt * Cc * 2);        // 32 MB
  __bf16* wqb  = (__bf16*)p; p += al((size_t)3 * Cc * Cc * 2);         // 24 MB
  __bf16* wob  = (__bf16*)p; p += al((size_t)Cc * Cc * 2);             //  8 MB
  __bf16* qkvb = (__bf16*)p; p += al((size_t)Bb * Tt * 3 * Cc * 2);    // 96 MB
  __bf16* Qb   = (__bf16*)p; p += al((size_t)Bb * Hh * Tt * HD * 2);   // 32 MB
  __bf16* Kb   = (__bf16*)p; p += al((size_t)Bb * Hh * Tt * HD * 2);   // 32 MB
  __bf16* Vb   = (__bf16*)p; p += al((size_t)Bb * Hh * Tt * HD * 2);   // 32 MB
  __bf16* Yb   = (__bf16*)p; p += al((size_t)Bb * Tt * Cc * 2);        // 32 MB

  const int nx = Bb * Tt * Cc;
  cvt_f32_to_bf16<<<nx / 4 / 256, 256, 0, stream>>>(x, xb, nx / 4);
  const int nq = 3 * Cc * Cc;
  cvt_f32_to_bf16<<<nq / 4 / 256, 256, 0, stream>>>(Wqkv, wqb, nq / 4);
  const int no = Cc * Cc;
  cvt_f32_to_bf16<<<no / 4 / 256, 256, 0, stream>>>(Wo, wob, no / 4);

  // qkv = x * Wqkv^T   (M=8192, N=6144, K=2048), bf16 out
  gemm_bf16_nt<<<dim3(3 * Cc / 128, Bb * Tt / 128), 256, 0, stream>>>(
      xb, wqb, qkvb, nullptr, Bb * Tt, 3 * Cc, Cc);

  const int npairs = Bb * Tt * 3 * Hh * (HD / 2);
  rope_split<<<npairs / 256, 256, 0, stream>>>(qkvb, cosT, sinT, Qb, Kb, Vb);

  flash_attn<<<Bb * Hh * (Tt / 128), 256, 0, stream>>>(Qb, Kb, Vb, Yb);

  // out = y * Wo^T   (M=8192, N=2048, K=2048), f32 out
  gemm_bf16_nt<<<dim3(Cc / 128, Bb * Tt / 128), 256, 0, stream>>>(
      Yb, wob, nullptr, out, Bb * Tt, Cc, Cc);
}
